// AdditiveAttention_39118562132543
// MI455X (gfx1250) — matrix operand order repaired and checked by reading, not compiled, not run on hardware
//
#include <hip/hip_runtime.h>
#include <hip/hip_bf16.h>

#define BB 64
#define SS 2048
#define DH 1024
#define DE 1024
#define DA 1024

#define ROWS 64            // encoder rows per workgroup (4 row-tiles of 16)
#define LDS_BYTES (ROWS * 1024 * 2)   // 128 KB bf16 tile

typedef __bf16 v16bf __attribute__((ext_vector_type(16)));
typedef float  v8f   __attribute__((ext_vector_type(8)));

union Frag {
    v16bf v;
    uint4 q[2];
};

__device__ __forceinline__ unsigned short f32_bf16(float f) {
    unsigned int u = __float_as_uint(f);
    return (unsigned short)((u + 0x7FFFu + ((u >> 16) & 1u)) >> 16);
}

// ---------------------------------------------------------------------------
// Kernel 1a: transpose + convert w_key [k][a] fp32 -> wkT [a][k] bf16 (2 MB)
// ---------------------------------------------------------------------------
__global__ __launch_bounds__(256) void wkey_transpose_bf16(
    const float* __restrict__ wk, unsigned short* __restrict__ wkT) {
    __shared__ float tile[16][17];
    const int a0 = blockIdx.x * 16, k0 = blockIdx.y * 16;
    const int tx = threadIdx.x, ty = threadIdx.y;
    tile[ty][tx] = wk[(size_t)(k0 + ty) * DA + (a0 + tx)];
    __syncthreads();
    wkT[(size_t)(a0 + ty) * DE + (k0 + tx)] = f32_bf16(tile[tx][ty]);
}

// ---------------------------------------------------------------------------
// Kernel 1b: query[b][a] = sum_d dh[b][d] * wq[d][a]   (134 MFLOP, tiny)
// ---------------------------------------------------------------------------
__global__ __launch_bounds__(256) void query_kernel(
    const float* __restrict__ dh, const float* __restrict__ wq,
    float* __restrict__ q) {
    const int b = blockIdx.x;
    const int a = blockIdx.y * 256 + threadIdx.x;
    float acc = 0.f;
    for (int d = 0; d < DH; ++d)
        acc += dh[(size_t)b * DH + d] * wq[(size_t)d * DA + a];
    q[(size_t)b * DA + a] = acc;
}

// ---------------------------------------------------------------------------
// Kernel 2: fused  keys = enc @ w_key ; scores = tanh(q + keys) . w_score
// One workgroup: ROWS=64 encoder rows (4 row-tiles) x all 1024 a-cols.
// 8 waves, each owns 128 a-cols (8 col-tiles). K loop over DE in steps of 32.
// B-operand (wkT, L2-resident) is reused across 4 row-tiles per wave,
// cutting L2 traffic to ~256 B per v_wmma.
// ---------------------------------------------------------------------------
__global__ __launch_bounds__(256) void keys_scores_kernel(
    const float* __restrict__ enc, const unsigned short* __restrict__ wkT,
    const float* __restrict__ q, const float* __restrict__ ws,
    const unsigned char* __restrict__ mask, float* __restrict__ scores) {
    // ROWS x 1024 bf16, XOR-swizzled in 8-element chunks (dynamic, 128 KB)
    extern __shared__ unsigned short ldsA[];

    const int tid = threadIdx.x;
    const int b  = blockIdx.x >> 5;           // 32 s-blocks per batch row
    const int s0 = (blockIdx.x & 31) * ROWS;  // 64 rows per block

    // ---- cooperative load: fp32 -> bf16, swizzle chunk ^ (row & 15) ----
    {
        const float* base = enc + ((size_t)b * SS + s0) * DE;
        #pragma unroll
        for (int i = 0; i < (ROWS * 1024 / 4) / 256; ++i) {
            const int idx  = tid + i * 256;   // float4 index
            const int flat = idx * 4;
            const int row  = flat >> 10;
            const int col  = flat & 1023;
            float4 f = *(const float4*)(base + (size_t)row * DE + col);
            const int chunk = col >> 3;
            const int sw    = (((chunk ^ (row & 15)) << 3) | (col & 7));
            unsigned short* dst = &ldsA[row * 1024 + sw];
            uint2 p;
            p.x = ((unsigned)f32_bf16(f.y) << 16) | f32_bf16(f.x);
            p.y = ((unsigned)f32_bf16(f.w) << 16) | f32_bf16(f.z);
            *(uint2*)dst = p;
        }
    }
    __syncthreads();

    const int lane = tid & 31;
    const int wv   = tid >> 5;      // wave id 0..7
    const int n    = lane & 15;     // N / M index within tile
    const int h    = lane >> 4;     // lane half
    const int colbase = wv << 7;    // 128 a-cols per wave

    float qv[8], wsv[8];
    #pragma unroll
    for (int ct = 0; ct < 8; ++ct) {
        const int a = colbase + ct * 16 + n;
        qv[ct]  = q[(size_t)b * DA + a];
        wsv[ct] = ws[a];
    }

    const v8f vzero = {0.f, 0.f, 0.f, 0.f, 0.f, 0.f, 0.f, 0.f};
    v8f acc[4][8];
    #pragma unroll
    for (int rt = 0; rt < 4; ++rt)
        #pragma unroll
        for (int ct = 0; ct < 8; ++ct) acc[rt][ct] = vzero;

    // ---- K loop: D = A(16x32 bf16) x B(32x16 bf16) + C, accumulate ----
    for (int k = 0; k < DE; k += 32) {
        Frag afr[4];
        #pragma unroll
        for (int rt = 0; rt < 4; ++rt) {
            const int row = rt * 16 + n;
            const int c0 = (k >> 3) + h;      // chunk of elems 0..7  (K = k+8h)
            const int c1 = c0 + 2;            // chunk of elems 8..15 (K = k+16+8h)
            afr[rt].q[0] = *(const uint4*)&ldsA[row * 1024 + ((c0 ^ n) << 3)];
            afr[rt].q[1] = *(const uint4*)&ldsA[row * 1024 + ((c1 ^ n) << 3)];
        }
        #pragma unroll
        for (int ct = 0; ct < 8; ++ct) {
            Frag bfr;
            const unsigned short* bp =
                wkT + (size_t)(colbase + ct * 16 + n) * DE + k + 8 * h;
            bfr.q[0] = *(const uint4*)bp;
            bfr.q[1] = *(const uint4*)(bp + 16);
            #pragma unroll
            for (int rt = 0; rt < 4; ++rt) {
                acc[rt][ct] = __builtin_amdgcn_wmma_f32_16x16x32_bf16(
                    false, afr[rt].v, false, bfr.v, (short)0, acc[rt][ct],
                    false, false);
            }
        }
    }

    // ---- epilogue: scores[row] = sum_a tanh(q[a] + keys[row][a]) * ws[a] ----
    __syncthreads();
    float* part = (float*)ldsA;     // reuse tile: [8 waves][ROWS]
    #pragma unroll
    for (int rt = 0; rt < 4; ++rt) {
        #pragma unroll
        for (int r = 0; r < 8; ++r) {
            float p = 0.f;
            #pragma unroll
            for (int ct = 0; ct < 8; ++ct)
                p += tanhf(qv[ct] + acc[rt][ct][r]) * wsv[ct];
            // reduce over the 16 lanes (16 a-values) in this half
            p += __shfl_xor(p, 1, 32);
            p += __shfl_xor(p, 2, 32);
            p += __shfl_xor(p, 4, 32);
            p += __shfl_xor(p, 8, 32);
            if (n == 0) {
                const int row = rt * 16 + r + (h << 3);   // M = r + 8*half
                part[wv * ROWS + row] = p;
            }
        }
    }
    __syncthreads();
    if (tid < ROWS) {
        float ssum = 0.f;
        #pragma unroll
        for (int w = 0; w < 8; ++w) ssum += part[w * ROWS + tid];  // fixed order
        const int sidx = s0 + tid;
        const float val = mask[(size_t)b * SS + sidx] ? ssum : -1e9f;
        scores[(size_t)b * SS + sidx] = val;
    }
}

// ---------------------------------------------------------------------------
// Kernel 3: softmax over S per batch row
// ---------------------------------------------------------------------------
__global__ __launch_bounds__(256) void softmax_kernel(
    const float* __restrict__ scores, float* __restrict__ weights) {
    __shared__ float red[256];
    const int b = blockIdx.x, tid = threadIdx.x;
    const float* srow = scores + (size_t)b * SS;
    float x[8];
    float mx = -3.4e38f;
    #pragma unroll
    for (int i = 0; i < 8; ++i) { x[i] = srow[tid + i * 256]; mx = fmaxf(mx, x[i]); }
    red[tid] = mx; __syncthreads();
    for (int off = 128; off > 0; off >>= 1) {
        if (tid < off) red[tid] = fmaxf(red[tid], red[tid + off]);
        __syncthreads();
    }
    mx = red[0]; __syncthreads();
    float sum = 0.f;
    #pragma unroll
    for (int i = 0; i < 8; ++i) { x[i] = expf(x[i] - mx); sum += x[i]; }
    red[tid] = sum; __syncthreads();
    for (int off = 128; off > 0; off >>= 1) {
        if (tid < off) red[tid] += red[tid + off];
        __syncthreads();
    }
    const float inv = 1.f / red[0];
    float* wrow = weights + (size_t)b * SS;
    #pragma unroll
    for (int i = 0; i < 8; ++i) wrow[tid + i * 256] = x[i] * inv;
}

// ---------------------------------------------------------------------------
// Kernel 4: partial context over s-chunks (deterministic two-stage reduce)
// ---------------------------------------------------------------------------
__global__ __launch_bounds__(256) void context_partial_kernel(
    const float* __restrict__ enc, const float* __restrict__ weights,
    float* __restrict__ partial) {
    const int b = blockIdx.x, ch = blockIdx.y, tid = threadIdx.x;
    const int d  = tid * 4;
    const int s0 = ch * 256;
    float4 acc = {0.f, 0.f, 0.f, 0.f};
    const float* base = enc + ((size_t)b * SS + s0) * DE + d;
    const float* wrow = weights + (size_t)b * SS + s0;
    for (int s = 0; s < 256; ++s) {
        const float w = wrow[s];
        float4 e = *(const float4*)(base + (size_t)s * DE);
        acc.x += w * e.x; acc.y += w * e.y; acc.z += w * e.z; acc.w += w * e.w;
    }
    *(float4*)(partial + (((size_t)ch * BB + b) * DE + d)) = acc;
}

__global__ __launch_bounds__(256) void context_reduce_kernel(
    const float* __restrict__ partial, float* __restrict__ ctx) {
    const int i = blockIdx.x * 256 + threadIdx.x;   // B*DE = 65536
    float s = 0.f;
    #pragma unroll
    for (int c = 0; c < 8; ++c) s += partial[(size_t)c * BB * DE + i];
    ctx[i] = s;
}

// ---------------------------------------------------------------------------
extern "C" void kernel_launch(void* const* d_in, const int* in_sizes, int n_in,
                              void* d_out, int out_size, void* d_ws, size_t ws_size,
                              hipStream_t stream) {
    const float*         dec_h = (const float*)d_in[0];          // [B, DH]
    const float*         enc   = (const float*)d_in[1];          // [B, S, DE]
    const unsigned char* mask  = (const unsigned char*)d_in[2];  // [B, S] bool
    const float*         w_q   = (const float*)d_in[3];          // [DH, DA]
    const float*         w_k   = (const float*)d_in[4];          // [DE, DA]
    const float*         w_s   = (const float*)d_in[5];          // [DA]

    float* ctx     = (float*)d_out;                 // [B, DE]
    float* weights = ctx + (size_t)BB * DE;         // [B, S]

    // scratch layout
    char* wsb = (char*)d_ws;
    unsigned short* wkT  = (unsigned short*)wsb;                       // 2 MB
    float* qbuf          = (float*)(wsb + (1u << 21));                 // 256 KB
    float* scores        = (float*)(wsb + (1u << 21) + (1u << 18));    // 512 KB
    float* partial       = (float*)(wsb + (1u << 21) + (1u << 18) + (1u << 19)); // 2 MB

    // opt in to >64KB dynamic LDS (CDNA5 allows up to 320 KB per workgroup)
    (void)hipFuncSetAttribute((const void*)keys_scores_kernel,
                              hipFuncAttributeMaxDynamicSharedMemorySize,
                              LDS_BYTES);

    wkey_transpose_bf16<<<dim3(DA / 16, DE / 16), dim3(16, 16), 0, stream>>>(w_k, wkT);
    query_kernel<<<dim3(BB, DA / 256), 256, 0, stream>>>(dec_h, w_q, qbuf);
    keys_scores_kernel<<<BB * (SS / ROWS), 256, LDS_BYTES, stream>>>(
        enc, wkT, qbuf, w_s, mask, scores);
    softmax_kernel<<<BB, 256, 0, stream>>>(scores, weights);
    context_partial_kernel<<<dim3(BB, SS / 256), 256, 0, stream>>>(enc, weights, partial);
    context_reduce_kernel<<<BB * DE / 256, 256, 0, stream>>>(partial, ctx);
}
